// StandardAttention_38585986187726
// MI455X (gfx1250) — matrix operand order repaired and checked by reading, not compiled, not run on hardware
//
#include <hip/hip_runtime.h>
#include <hip/hip_bf16.h>

// ---------------------------------------------------------------------------
// Types for CDNA5 WMMA
// ---------------------------------------------------------------------------
typedef __attribute__((ext_vector_type(16))) _Float16 v16h;
typedef __attribute__((ext_vector_type(8)))  _Float16 v8h;
typedef __attribute__((ext_vector_type(4)))  _Float16 v4h;
typedef __attribute__((ext_vector_type(8)))  float    v8f;

#define WMMA_F16(a, b, c) \
  __builtin_amdgcn_wmma_f32_16x16x32_f16(false, (a), false, (b), (short)0, (c), false, false)

// Problem constants (from reference setup_inputs)
#define BB 2
#define TT 2048
#define CC 1024
#define HH 16
#define DD 64

// ---------------------------------------------------------------------------
// gfx1250 async global->LDS copy (ASYNCcnt-tracked). The low 32 bits of a
// generic pointer to LDS are the wave-relative LDS byte offset (flat-LDS
// aperture mapping truncates to addr[31:0]).
// ---------------------------------------------------------------------------
__device__ __forceinline__ void async_b128(void* lds, const void* gmem) {
  unsigned int  l = (unsigned int)(unsigned long long)(uintptr_t)lds;
  unsigned long long g = (unsigned long long)(uintptr_t)gmem;
  asm volatile("global_load_async_to_lds_b128 %0, %1, off"
               :: "v"(l), "v"(g) : "memory");
}
__device__ __forceinline__ void wait_async0() {
  asm volatile("s_wait_asynccnt 0x0" ::: "memory");
}

// ---------------------------------------------------------------------------
// Fragment loaders (layouts per CDNA5 ISA 7.12.2)
//  A (16x32, f16): lanes 0-15 -> M=lane, halves 0..7 = K ks..ks+7, 8..15 = K ks+16..ks+23
//                  lanes16-31 -> M=lane-16, halves 0..7 = K ks+8..15, 8..15 = K ks+24..31
//  B (32x16, f16): N = nbase + lane%16, K order same as A
// ---------------------------------------------------------------------------
__device__ __forceinline__ v16h load_frag_a(const _Float16* base, int ld, int mbase, int ks) {
  int lane = threadIdx.x & 31;
  int m  = mbase + (lane & 15);
  int kb = ks + ((lane & 16) ? 8 : 0);
  v8h lo = *(const v8h*)(base + m * ld + kb);
  v8h hi = *(const v8h*)(base + m * ld + kb + 16);
  return __builtin_shufflevector(lo, hi, 0,1,2,3,4,5,6,7,8,9,10,11,12,13,14,15);
}

__device__ __forceinline__ v16h load_frag_b(const _Float16* base, int ld, int nbase, int ks) {
  int lane = threadIdx.x & 31;
  int n  = nbase + (lane & 15);
  int kb = ks + ((lane & 16) ? 8 : 0);
  v8h lo = *(const v8h*)(base + n * ld + kb);
  v8h hi = *(const v8h*)(base + n * ld + kb + 16);
  return __builtin_shufflevector(lo, hi, 0,1,2,3,4,5,6,7,8,9,10,11,12,13,14,15);
}

// ---------------------------------------------------------------------------
// Kernel 0a: f32 -> f16 conversion (n divisible by 4)
// ---------------------------------------------------------------------------
__global__ __launch_bounds__(256)
void cvt_f32_to_f16(const float* __restrict__ in, _Float16* __restrict__ out, int n4) {
  int i = blockIdx.x * blockDim.x + threadIdx.x;
  if (i >= n4) return;
  const float4 v = *(const float4*)(in + 4 * (size_t)i);
  v4h o;
  o[0] = (_Float16)v.x; o[1] = (_Float16)v.y; o[2] = (_Float16)v.z; o[3] = (_Float16)v.w;
  *(v4h*)(out + 4 * (size_t)i) = o;
}

// ---------------------------------------------------------------------------
// Kernel 0b: f32 -> f16 conversion with transpose: in[K][N] -> out[N][K].
// One-time cost so GEMM LDS staging is fully contiguous b128.
// ---------------------------------------------------------------------------
__global__ __launch_bounds__(256)
void cvt_transpose_f32_f16(const float* __restrict__ in, _Float16* __restrict__ out,
                           int K, int N) {
  int i = blockIdx.x * blockDim.x + threadIdx.x;
  if (i >= K * N) return;
  int k = i / N, n = i % N;
  out[(size_t)n * K + k] = (_Float16)in[i];
}

// ---------------------------------------------------------------------------
// Kernel 1/4: tiled WMMA GEMM  C[M,N] = A[M,K] * Bt[N,K]^T + bias[N]
// A row-major f16 (M x K), Bt row-major f16 (N x K, i.e. B pre-transposed).
// Block: 256 threads = 8 waves (4x2), tile BM=128 x BN=64, BK=64.
// ---------------------------------------------------------------------------
#define GBM 128
#define GBN 64
#define GBK 64

__global__ __launch_bounds__(256)
void gemm_f16_f32(const _Float16* __restrict__ A, const _Float16* __restrict__ Btg,
                  const float* __restrict__ bias, float* __restrict__ C,
                  int M, int N, int K) {
  __shared__ __align__(16) _Float16 As[GBM][GBK];      // [m][k]
  __shared__ __align__(16) _Float16 Bs[GBN][GBK];      // [n][k]

  const int tid  = threadIdx.x;
  const int lane = tid & 31;
  const int wid  = tid >> 5;
  const int wm   = wid >> 1;          // 0..3
  const int wn   = wid & 1;           // 0..1
  const int bm   = blockIdx.y * GBM;
  const int bn   = blockIdx.x * GBN;

  v8f acc[2][2];
#pragma unroll
  for (int i = 0; i < 2; ++i)
#pragma unroll
    for (int j = 0; j < 2; ++j) acc[i][j] = (v8f)0.0f;

  const int a_row = tid >> 1;               // 0..127
  const int a_off = (tid & 1) * 32;         // 0 or 32 halves
  const int b_row = tid >> 2;               // 0..63
  const int b_off = (tid & 3) * 16;         // 0,16,32,48 halves

  const _Float16* srcA = A   + (size_t)(bm + a_row) * K + a_off;
  const _Float16* srcB = Btg + (size_t)(bn + b_row) * K + b_off;

  for (int k0 = 0; k0 < K; k0 += GBK) {
    // ---- async-stage A (128x64) and B^T (64x64) tiles, contiguous b128s
#pragma unroll
    for (int u = 0; u < 4; ++u)
      async_b128(&As[a_row][a_off + 8 * u], srcA + k0 + 8 * u);
#pragma unroll
    for (int u = 0; u < 2; ++u)
      async_b128(&Bs[b_row][b_off + 8 * u], srcB + k0 + 8 * u);
    if (k0 + GBK < K) {   // prefetch next k-tile into GL2
      __builtin_prefetch(srcA + k0 + GBK, 0, 1);
      __builtin_prefetch(srcB + k0 + GBK, 0, 1);
    }
    wait_async0();
    __syncthreads();

#pragma unroll
    for (int ks = 0; ks < GBK; ks += 32) {
      v16h a0 = load_frag_a(&As[0][0], GBK, wm * 32,      ks);
      v16h a1 = load_frag_a(&As[0][0], GBK, wm * 32 + 16, ks);
      v16h b0 = load_frag_b(&Bs[0][0], GBK, wn * 32,      ks);
      v16h b1 = load_frag_b(&Bs[0][0], GBK, wn * 32 + 16, ks);
      acc[0][0] = WMMA_F16(a0, b0, acc[0][0]);
      acc[0][1] = WMMA_F16(a0, b1, acc[0][1]);
      acc[1][0] = WMMA_F16(a1, b0, acc[1][0]);
      acc[1][1] = WMMA_F16(a1, b1, acc[1][1]);
    }
    __syncthreads();
  }

  // ---- epilogue: C/D layout: vgpr r -> rows r (lanes 0-15) and r+8 (lanes 16-31)
  const int rlo = (lane & 16) ? 8 : 0;
#pragma unroll
  for (int ti = 0; ti < 2; ++ti) {
#pragma unroll
    for (int tj = 0; tj < 2; ++tj) {
      const int mb  = bm + wm * 32 + ti * 16;
      const int col = bn + wn * 32 + tj * 16 + (lane & 15);
      const float bv = bias[col];
#pragma unroll
      for (int r = 0; r < 8; ++r)
        C[(size_t)(mb + r + rlo) * N + col] = acc[ti][tj][r] + bv;
    }
  }
}

// ---------------------------------------------------------------------------
// Kernel 2: RMS-norm + head-indexed rotary. One wave (32 lanes) per 64-wide row.
// cls 0 = q (norm+rope) -> (B,H,T,D); cls 1 = k (norm+rope) -> (B,H,T,D);
// cls 2 = v (plain convert) -> TRANSPOSED (B,H,D,T) so flash stages V^T with b128.
// ---------------------------------------------------------------------------
__global__ __launch_bounds__(256)
void rms_rope(const float* __restrict__ qkv,
              _Float16* __restrict__ Qo, _Float16* __restrict__ Ko,
              _Float16* __restrict__ Vt) {
  const int BHT = BB * HH * TT;
  const int lane = threadIdx.x & 31;
  const int row  = blockIdx.x * 8 + (threadIdx.x >> 5);
  const int cls  = row / BHT;
  const int r2   = row % BHT;
  const int b = r2 / (HH * TT);
  const int h = (r2 / TT) % HH;
  const int t = r2 % TT;

  const float* src = qkv + ((size_t)(b * TT + t)) * (3 * CC) + cls * CC + h * DD;
  const float e0 = src[lane];
  const float e1 = src[lane + 32];

  if (cls == 2) {
    const size_t vbase = ((size_t)(b * HH + h) * DD) * TT + t;
    Vt[vbase + (size_t)lane * TT]        = (_Float16)e0;
    Vt[vbase + (size_t)(lane + 32) * TT] = (_Float16)e1;
    return;
  }
  float ss = e0 * e0 + e1 * e1;
#pragma unroll
  for (int off = 16; off > 0; off >>= 1) ss += __shfl_xor(ss, off);
  const float rn = rsqrtf(ss * (1.0f / 64.0f) + 1.1920929e-07f);
  const float n0 = e0 * rn, n1 = e1 * rn;

  // inv_freq = 10000^(-lane/32); angle indexed by HEAD (faithful to reference bug)
  const float inv_freq = __expf(-(float)lane * (9.2103403719761836f / 32.0f));
  const float ang = (float)h * inv_freq;
  float c, s;
  __sincosf(ang, &c, &s);
  const float y0 =  n0 * c + n1 * s;
  const float y1 = -n0 * s + n1 * c;

  _Float16* out = (cls == 0) ? Qo : Ko;
  const size_t dst = ((size_t)(b * HH + h) * TT + t) * DD;
  out[dst + lane]      = (_Float16)y0;
  out[dst + lane + 32] = (_Float16)y1;
}

// ---------------------------------------------------------------------------
// Kernel 3: causal flash attention with WMMA.
// Block = 128 threads (4 waves); 64 query rows of one (b,h) per block; each
// wave owns 16 query rows. Key tiles of 64, online softmax.
// Q,K in (B,H,T,D); V pre-transposed (B,H,D,T). O written (B,T,H*D) f16.
// ---------------------------------------------------------------------------
__global__ __launch_bounds__(128)
void flash_attn(const _Float16* __restrict__ Q, const _Float16* __restrict__ K,
                const _Float16* __restrict__ Vt, _Float16* __restrict__ O) {
  __shared__ __align__(16) _Float16 qs[64][64];       // [q][d]
  __shared__ __align__(16) _Float16 kt[64][64];       // [key][d]  (B^T for QK^T)
  __shared__ __align__(16) _Float16 vs[64][64];       // [d][key]  (B^T for P@V)
  __shared__ __align__(16) _Float16 ps[4][16][64];    // per-wave P tile [q][key]

  const int tid  = threadIdx.x;
  const int lane = tid & 31;
  const int w    = tid >> 5;
  const int qt   = blockIdx.x;          // query tile (64 rows)
  const int h    = blockIdx.y;
  const int b    = blockIdx.z;
  const int qbase = qt * 64;

  const size_t head_off = ((size_t)(b * HH + h)) * TT * DD;
  const _Float16* Qp  = Q  + head_off;
  const _Float16* Kp  = K  + head_off;
  const _Float16* Vtp = Vt + head_off;   // (D, T) per head

  const int srow = tid >> 1;             // 0..63
  const int soff = (tid & 1) * 32;       // 0 or 32 halves

  // ---- stage Q tile once (async), pull per-wave A fragments into registers
#pragma unroll
  for (int u = 0; u < 4; ++u)
    async_b128(&qs[srow][soff + 8 * u],
               Qp + (size_t)(qbase + srow) * DD + soff + 8 * u);
  wait_async0();
  __syncthreads();
  const v16h qf0 = load_frag_a(&qs[0][0], 64, w * 16, 0);
  const v16h qf1 = load_frag_a(&qs[0][0], 64, w * 16, 32);

  float m[8], l[8];
  v8f o[4];
#pragma unroll
  for (int r = 0; r < 8; ++r) { m[r] = -1e30f; l[r] = 0.0f; }
#pragma unroll
  for (int nt = 0; nt < 4; ++nt) o[nt] = (v8f)0.0f;

  const float scale = 0.125f;  // 1/sqrt(64)
  const int rlo = (lane & 16) ? 8 : 0;

  for (int j = 0; j <= qt; ++j) {
    const int kb0 = j * 64;
    __syncthreads();   // previous tile's compute done before overwrite
    // ---- async-stage K tile [key][d] and V^T tile [d][key]; all contiguous
#pragma unroll
    for (int u = 0; u < 4; ++u)
      async_b128(&kt[srow][soff + 8 * u],
                 Kp + (size_t)(kb0 + srow) * DD + soff + 8 * u);
#pragma unroll
    for (int u = 0; u < 4; ++u)
      async_b128(&vs[srow][soff + 8 * u],
                 Vtp + (size_t)srow * TT + kb0 + soff + 8 * u);
    if (j < qt) {  // prefetch next key tile into GL2
      __builtin_prefetch(Kp + (size_t)(kb0 + 64 + srow) * DD + soff, 0, 1);
      __builtin_prefetch(Vtp + (size_t)srow * TT + kb0 + 64 + soff, 0, 1);
    }
    wait_async0();
    __syncthreads();

    // ---- S = Q K^T (16 rows x 64 keys per wave)
    v8f s[4];
#pragma unroll
    for (int nt = 0; nt < 4; ++nt) {
      s[nt] = (v8f)0.0f;
      v16h bk0 = load_frag_b(&kt[0][0], 64, nt * 16, 0);
      v16h bk1 = load_frag_b(&kt[0][0], 64, nt * 16, 32);
      s[nt] = WMMA_F16(qf0, bk0, s[nt]);
      s[nt] = WMMA_F16(qf1, bk1, s[nt]);
    }

    // ---- scale + causal mask (diagonal tile only)
    const bool diag = (j == qt);
#pragma unroll
    for (int nt = 0; nt < 4; ++nt)
#pragma unroll
      for (int r = 0; r < 8; ++r) {
        float v = s[nt][r] * scale;
        if (diag) {
          const int qrow = qbase + w * 16 + r + rlo;
          const int kcol = kb0 + nt * 16 + (lane & 15);
          if (kcol > qrow) v = -1e30f;
        }
        s[nt][r] = v;
      }

    // ---- online softmax stats (rows striped across 16-lane halves)
    float mt[8];
#pragma unroll
    for (int r = 0; r < 8; ++r) {
      mt[r] = fmaxf(fmaxf(s[0][r], s[1][r]), fmaxf(s[2][r], s[3][r]));
#pragma unroll
      for (int off = 8; off > 0; off >>= 1) mt[r] = fmaxf(mt[r], __shfl_xor(mt[r], off));
    }
    float al[8], ls[8];
#pragma unroll
    for (int r = 0; r < 8; ++r) {
      const float mn = fmaxf(m[r], mt[r]);
      al[r] = __expf(m[r] - mn);
      m[r]  = mn;
      ls[r] = 0.0f;
    }
#pragma unroll
    for (int nt = 0; nt < 4; ++nt)
#pragma unroll
      for (int r = 0; r < 8; ++r) {
        const float p = __expf(s[nt][r] - m[r]);
        s[nt][r] = p;
        ls[r] += p;
      }
#pragma unroll
    for (int r = 0; r < 8; ++r) {
#pragma unroll
      for (int off = 8; off > 0; off >>= 1) ls[r] += __shfl_xor(ls[r], off);
      l[r] = l[r] * al[r] + ls[r];
    }
#pragma unroll
    for (int nt = 0; nt < 4; ++nt)
#pragma unroll
      for (int r = 0; r < 8; ++r) o[nt][r] *= al[r];

    // ---- P (C layout) -> per-wave LDS -> A fragments
#pragma unroll
    for (int nt = 0; nt < 4; ++nt)
#pragma unroll
      for (int r = 0; r < 8; ++r)
        ps[w][r + rlo][nt * 16 + (lane & 15)] = (_Float16)s[nt][r];

    // ---- O += P @ V
    const v16h pf0 = load_frag_a(&ps[w][0][0], 64, 0, 0);
    const v16h pf1 = load_frag_a(&ps[w][0][0], 64, 0, 32);
#pragma unroll
    for (int nt = 0; nt < 4; ++nt) {
      v16h bv0 = load_frag_b(&vs[0][0], 64, nt * 16, 0);
      v16h bv1 = load_frag_b(&vs[0][0], 64, nt * 16, 32);
      o[nt] = WMMA_F16(pf0, bv0, o[nt]);
      o[nt] = WMMA_F16(pf1, bv1, o[nt]);
    }
  }

  // ---- normalize and store to (B, T, H*D) f16
#pragma unroll
  for (int nt = 0; nt < 4; ++nt)
#pragma unroll
    for (int r = 0; r < 8; ++r) {
      const int row = qbase + w * 16 + r + rlo;
      const int d   = nt * 16 + (lane & 15);
      O[((size_t)(b * TT + row)) * CC + h * DD + d] = (_Float16)(o[nt][r] / l[r]);
    }
}

// ---------------------------------------------------------------------------
// Host-side launch
// ---------------------------------------------------------------------------
extern "C" void kernel_launch(void* const* d_in, const int* in_sizes, int n_in,
                              void* d_out, int out_size, void* d_ws, size_t ws_size,
                              hipStream_t stream) {
  const float* x  = (const float*)d_in[0];   // (B,T,C)
  const float* Wa = (const float*)d_in[1];   // (C,3C)
  const float* ba = (const float*)d_in[2];   // (3C,)
  const float* Wp = (const float*)d_in[3];   // (C,C)
  const float* bp = (const float*)d_in[4];   // (C,)
  float* out = (float*)d_out;                // (B,T,C) f32

  const int M  = BB * TT;      // 4096
  const int N1 = 3 * CC;       // 3072
  const int Kc = CC;           // 1024

  char* ws = (char*)d_ws;
  size_t off = 0;
  auto alloc = [&](size_t bytes) -> void* {
    void* p = ws + off;
    off = (off + bytes + 255) & ~(size_t)255;
    return p;
  };
  _Float16* x16   = (_Float16*)alloc((size_t)M * Kc * 2);
  _Float16* waT   = (_Float16*)alloc((size_t)N1 * Kc * 2);  // (3C, C) = W_attn^T
  _Float16* wpT   = (_Float16*)alloc((size_t)CC * Kc * 2);  // (C, C)  = W_proj^T
  float*    qkv32 = (float*)   alloc((size_t)M * N1 * 4);
  _Float16* q16   = (_Float16*)alloc((size_t)M * CC * 2);   // (B,H,T,D)
  _Float16* k16   = (_Float16*)alloc((size_t)M * CC * 2);   // (B,H,T,D)
  _Float16* v16t  = (_Float16*)alloc((size_t)M * CC * 2);   // (B,H,D,T)
  _Float16* o16   = (_Float16*)alloc((size_t)M * CC * 2);   // (B,T,C)

  // 0) convert inputs to f16 (weights transposed to N x K)
  {
    int n4 = (M * Kc) / 4;
    cvt_f32_to_f16<<<dim3((n4 + 255) / 256), dim3(256), 0, stream>>>(x, x16, n4);
    int n = Kc * N1;
    cvt_transpose_f32_f16<<<dim3((n + 255) / 256), dim3(256), 0, stream>>>(Wa, waT, Kc, N1);
    n = Kc * CC;
    cvt_transpose_f32_f16<<<dim3((n + 255) / 256), dim3(256), 0, stream>>>(Wp, wpT, Kc, CC);
  }

  // 1) qkv = x @ W_attn + b_attn
  gemm_f16_f32<<<dim3(N1 / GBN, M / GBM), dim3(256), 0, stream>>>(
      x16, waT, ba, qkv32, M, N1, Kc);

  // 2) RMS-norm + rotary -> Q/K f16 (B,H,T,D); V -> f16 transposed (B,H,D,T)
  {
    const int rows = 3 * BB * HH * TT;
    rms_rope<<<dim3(rows / 8), dim3(256), 0, stream>>>(qkv32, q16, k16, v16t);
  }

  // 3) causal flash attention -> O (B,T,C) f16
  flash_attn<<<dim3(TT / 64, HH, BB), dim3(128), 0, stream>>>(q16, k16, v16t, o16);

  // 4) out = O @ W_proj + b_proj
  gemm_f16_f32<<<dim3(CC / GBN, M / GBM), dim3(256), 0, stream>>>(
      o16, wpT, bp, out, M, CC, Kc);
}
